// Cross_Guide_Fusion_35424890257929
// MI455X (gfx1250) — matrix operand order repaired and checked by reading, not compiled, not run on hardware
//
#include <hip/hip_runtime.h>
#include <hip/hip_bf16.h>
#include <math.h>

// ---------------------------------------------------------------------------
// CDNA5 (gfx1250, wave32) implementation of Cross_Guide_Fusion.
// All matrix contractions use v_wmma_f32_16x16x32_bf16 (bf16 in, f32 acc).
// Weights are pre-converted to bf16 in fragment-friendly layouts so operand
// loads are 128-bit vector loads (global_load_b128 / ds_load_b128).
// ---------------------------------------------------------------------------

typedef __attribute__((ext_vector_type(16))) __bf16 v16bf;
typedef __attribute__((ext_vector_type(8)))  __bf16 v8bf;
typedef __attribute__((ext_vector_type(8)))  float  v8f;

#define WMMA_BF16(A_, B_, C_) \
  __builtin_amdgcn_wmma_f32_16x16x32_bf16(false, (A_), false, (B_), (short)0, (C_), false, false)

#define NSPAT  65536            // W*H per batch
#define NTOT   262144           // B * NSPAT  (rows of the (b,n,c) matrices)

// fragment index helpers (wave32, 16-bit): akk = ISA table k order, bkk = contiguous k order
static __device__ __forceinline__ int akk(int lane, int e) {
  // A[m][k]: m = lane&15 ; k = kbase + e (+8 for upper half), kbase = lane<16?0:8
  return ((lane < 16) ? 0 : 8) + e + ((e >= 8) ? 8 : 0);
}
static __device__ __forceinline__ int bkk(int lane, int e) {
  // B[k][n]: n = lane&15 ; k = kb2 + e, kb2 = lane<16?0:16
  return ((lane < 16) ? 0 : 16) + e;
}

// A fragment = two contiguous 8x bf16 chunks at p0 and p1
static __device__ __forceinline__ v16bf frag2(const __bf16* p0, const __bf16* p1) {
  v8bf lo = *(const v8bf*)p0;
  v8bf hi = *(const v8bf*)p1;
  return __builtin_shufflevector(lo, hi, 0, 1, 2, 3, 4, 5, 6, 7,
                                 8, 9, 10, 11, 12, 13, 14, 15);
}

static __device__ __forceinline__ float gelu_exact(float x) {
  return 0.5f * x * (1.0f + erff(x * 0.70710678118654752f));
}

// ---------------------------------------------------------------------------
// Weight prep: f32 -> bf16, optionally 64x64 transpose (to [n][k] layout)
// ---------------------------------------------------------------------------
__global__ void cvt_copy_kernel(const float* __restrict__ src,
                                __bf16* __restrict__ dst, int n) {
  int i = blockIdx.x * 256 + threadIdx.x;
  if (i < n) dst[i] = (__bf16)src[i];
}
__global__ void cvt_t64_kernel(const float* __restrict__ src,
                               __bf16* __restrict__ dst) {
  int i = blockIdx.x * 256 + threadIdx.x;     // 4096
  int nn = i >> 6, k = i & 63;
  dst[i] = (__bf16)src[k * 64 + nn];          // dst[n][k] = src[k][n]
}

// ---------------------------------------------------------------------------
// K1: transpose + layernorm over channels. in (b,c,w,h) f32 ->
//     optional f32 out in same (b,c,w,h) layout, bf16 out in (b, n=ih*W+iw, c).
// ---------------------------------------------------------------------------
__global__ void ln_tr_kernel(const float* __restrict__ in,
                             const float* __restrict__ g,
                             const float* __restrict__ bta,
                             float* __restrict__ outF,      // may be nullptr
                             __bf16* __restrict__ outB) {
  int p = blockIdx.x * 256 + threadIdx.x;          // [0, NTOT)
  int b = p >> 16, s = p & 65535;                  // s = iw*256 + ih
  size_t base = ((size_t)(b * 64)) << 16;
  float sum = 0.f, sq = 0.f;
  for (int c = 0; c < 64; ++c) {
    float v = in[base + ((size_t)c << 16) + s];
    sum += v; sq += v * v;
  }
  float mean = sum * (1.0f / 64.0f);
  float var  = sq * (1.0f / 64.0f) - mean * mean;
  float inv  = rsqrtf(var + 1e-5f);
  int n = ((s & 255) << 8) | (s >> 8);             // n = ih*256 + iw
  size_t rowoff = (size_t)((b << 16) | n) * 64;
  for (int c = 0; c < 64; ++c) {
    float v = (in[base + ((size_t)c << 16) + s] - mean) * inv * g[c] + bta[c];
    if (outF) outF[base + ((size_t)c << 16) + s] = v;
    outB[rowoff + c] = (__bf16)v;
  }
}

// ---------------------------------------------------------------------------
// K2: [rows,64] x [64,64] GEMM. A bf16 [row][k]; Wt bf16 [n][k].
//     epi: 0 -> bf16 out [row,64]; 1 -> f32 out [row,64];
//          2 -> ffn2 epilogue: d_out NCHW = val + resid[row,col]
//     act: 0 none, 2 leaky(0.1)
// ---------------------------------------------------------------------------
__global__ void gemm_n64_kernel(const __bf16* __restrict__ A,
                                const __bf16* __restrict__ Wt,
                                int act, int epi,
                                float* __restrict__ outF,
                                __bf16* __restrict__ outB,
                                const float* __restrict__ resid,
                                float* __restrict__ outNCHW) {
  const int lane = threadIdx.x & 31;
  const int wave = threadIdx.x >> 5;
  const int row0 = (blockIdx.x * 4 + wave) * 16;
  const int m    = lane & 15;
  const int kbA  = (lane < 16) ? 0 : 8;
  const int kbB  = (lane < 16) ? 0 : 16;

  v8f acc[4] = {};
  for (int ks = 0; ks < 2; ++ks) {
    const __bf16* Ap = A + (size_t)(row0 + m) * 64 + ks * 32 + kbA;
    v16bf a = frag2(Ap, Ap + 16);
#pragma unroll
    for (int t = 0; t < 4; ++t) {
      const __bf16* Bp = Wt + (t * 16 + m) * 64 + ks * 32 + kbA;
      v16bf bb = frag2(Bp, Bp + 16);
      acc[t] = WMMA_BF16(a, bb, acc[t]);
    }
  }
#pragma unroll
  for (int t = 0; t < 4; ++t) {
#pragma unroll
    for (int r = 0; r < 8; ++r) {
      int mm  = ((lane < 16) ? 0 : 8) + r;
      int row = row0 + mm;
      int col = t * 16 + m;
      float v = acc[t][r];
      if (act == 2) v = (v > 0.f) ? v : 0.1f * v;
      if (epi == 0) {
        outB[(size_t)row * 64 + col] = (__bf16)v;
      } else if (epi == 1) {
        outF[(size_t)row * 64 + col] = v;
      } else {
        int b = row >> 16, n = row & 65535;
        int iw = n & 255, ih = n >> 8;
        float res = resid[(size_t)row * 64 + col];
        outNCHW[(size_t)(((b * 64 + col) << 16)) + iw * 256 + ih] = v + res;
      }
    }
  }
}

// ---------------------------------------------------------------------------
// K3: per-column sum of squares over n (for folded L2 norms of k and q)
// ---------------------------------------------------------------------------
__global__ void norms_kernel(const __bf16* __restrict__ kbf,
                             const __bf16* __restrict__ qbf,
                             float* __restrict__ nk2,
                             float* __restrict__ nq2) {
  __shared__ float red[256];
  int idx = blockIdx.x;
  int which = idx >> 8;                 // 0 -> k, 1 -> q
  int rem = idx & 255;
  int b = rem >> 6, c = rem & 63;
  const __bf16* src = which ? qbf : kbf;
  float s = 0.f;
  for (int i = threadIdx.x; i < NSPAT; i += 256) {
    float v = (float)src[(size_t)((b << 16) | i) * 64 + c];
    s += v * v;
  }
  red[threadIdx.x] = s;
  __syncthreads();
  for (int st = 128; st > 0; st >>= 1) {
    if (threadIdx.x < st) red[threadIdx.x] += red[threadIdx.x + st];
    __syncthreads();
  }
  if (threadIdx.x == 0) (which ? nq2 : nk2)[b * 64 + c] = red[0];
}

// ---------------------------------------------------------------------------
// K4: Gram matrices S[bh][d][e] = sum_n k[d,n] q[e,n] via bf16 WMMA partials
//     + f32 global atomic reduction. grid = (64 n-chunks, 16 bh), block 256.
// ---------------------------------------------------------------------------
__global__ void gram_kernel(const __bf16* __restrict__ kbf,
                            const __bf16* __restrict__ qbf,
                            float* __restrict__ S) {
  const int lane = threadIdx.x & 31;
  const int wave = threadIdx.x >> 5;
  const int bh = blockIdx.y;
  const int b = bh >> 2, h = bh & 3;
  const int m = lane & 15;
  const int n0 = blockIdx.x * 1024 + wave * 128;
  const size_t brow = (size_t)b << 16;

  v8f acc = {};
  for (int kb = 0; kb < 4; ++kb) {
    int nb = n0 + kb * 32;
    v16bf a, bb;
#pragma unroll
    for (int e = 0; e < 16; ++e)
      a[e] = kbf[(brow + nb + akk(lane, e)) * 64 + h * 16 + m];   // A[d][n] = k^T
#pragma unroll
    for (int e = 0; e < 16; ++e)
      bb[e] = qbf[(brow + nb + akk(lane, e)) * 64 + h * 16 + m];  // B[n][e] = q
    acc = WMMA_BF16(a, bb, acc);
  }
#pragma unroll
  for (int r = 0; r < 8; ++r) {
    int d = ((lane < 16) ? 0 : 8) + r;
    atomicAdd(&S[bh * 256 + d * 16 + m], acc[r]);
  }
}

// ---------------------------------------------------------------------------
// K5: fold L2 norms into Gram, softmax over e. one block, 256 threads.
// ---------------------------------------------------------------------------
__global__ void softmax_kernel(const float* __restrict__ S,
                               const float* __restrict__ nk2,
                               const float* __restrict__ nq2,
                               float* __restrict__ attn) {
  int t = threadIdx.x;
  int bh = t >> 4, d = t & 15;
  int b = bh >> 2, h = bh & 3;
  float nk = fmaxf(sqrtf(nk2[b * 64 + h * 16 + d]), 1e-12f);
  float row[16];
  float mx = -3.4e38f;
  for (int e = 0; e < 16; ++e) {
    float nq = fmaxf(sqrtf(nq2[b * 64 + h * 16 + e]), 1e-12f);
    row[e] = S[bh * 256 + d * 16 + e] / (nk * nq);
    mx = fmaxf(mx, row[e]);
  }
  float sum = 0.f;
  for (int e = 0; e < 16; ++e) { row[e] = expf(row[e] - mx); sum += row[e]; }
  float inv = 1.0f / sum;
  for (int e = 0; e < 16; ++e) attn[bh * 256 + d * 16 + e] = row[e] * inv;
}

// ---------------------------------------------------------------------------
// K6: xo = attn @ v (VALU, staged to LDS as bf16), then out_c =
//     gelu(xo @ w_proj + b_proj) via WMMA (A from LDS, b128 ds loads).
// ---------------------------------------------------------------------------
__global__ void av_proj_kernel(const __bf16* __restrict__ vbf,
                               const float* __restrict__ attnG,
                               const __bf16* __restrict__ wprojT,  // [n][k] bf16
                               const float* __restrict__ bproj,
                               float* __restrict__ srOut) {
  __shared__ float attn_s[1024];
  __shared__ __align__(16) __bf16 xo[4][1024];
  const int lane = threadIdx.x & 31;
  const int wave = threadIdx.x >> 5;
  const int b = blockIdx.x >> 10;                 // 64 rows per block, same batch
  for (int i = threadIdx.x; i < 1024; i += 128)
    attn_s[i] = attnG[b * 1024 + i];
  __syncthreads();

  const int row0 = (blockIdx.x * 4 + wave) * 16;
  for (int idx = lane; idx < 1024; idx += 32) {
    int rr = idx >> 6, col = idx & 63, h = col >> 4, d = col & 15;
    size_t rowoff = (size_t)(row0 + rr) * 64 + h * 16;
    float sum = 0.f;
#pragma unroll
    for (int e = 0; e < 16; ++e)
      sum += attn_s[h * 256 + d * 16 + e] * (float)vbf[rowoff + e];
    xo[wave][rr * 64 + col] = (__bf16)sum;
  }
  __syncthreads();

  const int m = lane & 15;
  const int kbA = (lane < 16) ? 0 : 8;
  const int kbB = (lane < 16) ? 0 : 16;
  v8f acc[4] = {};
  for (int ks = 0; ks < 2; ++ks) {
    const __bf16* Ap = &xo[wave][m * 64 + ks * 32 + kbA];
    v16bf a = frag2(Ap, Ap + 16);
#pragma unroll
    for (int t = 0; t < 4; ++t) {
      const __bf16* Bp = wprojT + (t * 16 + m) * 64 + ks * 32 + kbA;
      v16bf bb = frag2(Bp, Bp + 16);
      acc[t] = WMMA_BF16(a, bb, acc[t]);
    }
  }
#pragma unroll
  for (int t = 0; t < 4; ++t) {
#pragma unroll
    for (int r = 0; r < 8; ++r) {
      int mm = ((lane < 16) ? 0 : 8) + r;
      int col = t * 16 + m;
      float v = gelu_exact(acc[t][r] + bproj[col]);
      srOut[(size_t)(row0 + mm) * 64 + col] = v;
    }
  }
}

// ---------------------------------------------------------------------------
// K7: 3x3 conv (64->64, pad 1) as implicit GEMM with bf16 WMMA. K = 576.
//     Input halo tile staged in LDS (per wave: 64ci x 3 x 18 bf16).
//     mode 0: out = act(conv+bias) to f32 NCHW
//     mode 1: sr[row,c] += conv+bias + x_msi (completes sr = out_c+out_p+xm)
// ---------------------------------------------------------------------------
__global__ void conv3x3_kernel(const float* __restrict__ in,
                               const __bf16* __restrict__ Wbf,  // [64][576] bf16
                               const float* __restrict__ bias,
                               int act, int mode,
                               float* __restrict__ outNCHW,
                               float* __restrict__ srbuf,
                               const float* __restrict__ xmsi) {
  __shared__ __align__(16) __bf16 tileIn[4][3456];   // [ci*3+dy][xoff 0..17]
  const int lane = threadIdx.x & 31;
  const int wave = threadIdx.x >> 5;
  const int m = lane & 15;
  const int pbase = (blockIdx.x * 4 + wave) * 16;   // 16 contiguous ih pixels
  const int b = pbase >> 16, sbase = pbase & 65535;
  const int iwb = sbase >> 8, ihb = sbase & 255;
  const int s = sbase + m;                           // this lane's pixel

  // stage input halo (zero-padded) into LDS, coalesced along xoff
  for (int idx = lane; idx < 3456; idx += 32) {
    int xoff = idx % 18;
    int tmp  = idx / 18;          // ci*3 + dy
    int dy = tmp % 3, ci = tmp / 3;
    int yy = iwb + dy - 1;
    int xx = ihb + xoff - 1;
    float v = 0.f;
    if (yy >= 0 && yy < 256 && xx >= 0 && xx < 256)
      v = in[(size_t)(((b * 64 + ci) << 16)) + yy * 256 + xx];
    tileIn[wave][idx] = (__bf16)v;
  }
  __syncthreads();

  const int kbA = (lane < 16) ? 0 : 8;
  const int kbB = (lane < 16) ? 0 : 16;
  v8f acc[4] = {};
  for (int ks = 0; ks < 18; ++ks) {
    v16bf bb;
#pragma unroll
    for (int e = 0; e < 16; ++e) {
      int kk = ks * 32 + kbB + e;
      int ci = kk / 9, rem = kk - ci * 9;
      int dy = rem / 3, dx = rem - dy * 3;
      bb[e] = tileIn[wave][(ci * 3 + dy) * 18 + m + dx];
    }
#pragma unroll
    for (int t = 0; t < 4; ++t) {
      const __bf16* Ap = Wbf + (t * 16 + m) * 576 + ks * 32 + kbB;
      v16bf a = frag2(Ap, Ap + 8);
      acc[t] = WMMA_BF16(a, bb, acc[t]);
    }
  }
#pragma unroll
  for (int t = 0; t < 4; ++t) {
#pragma unroll
    for (int r = 0; r < 8; ++r) {
      int co = t * 16 + ((lane < 16) ? 0 : 8) + r;
      float v = acc[t][r] + bias[co];
      if (act == 1) v = gelu_exact(v);
      if (mode == 0) {
        outNCHW[(size_t)(((b * 64 + co) << 16)) + s] = v;
      } else {
        int n = ((s & 255) << 8) | (s >> 8);
        size_t i = (size_t)((b << 16) | n) * 64 + co;
        srbuf[i] = srbuf[i] + v + xmsi[(size_t)(((b * 64 + co) << 16)) + s];
      }
    }
  }
}

// ---------------------------------------------------------------------------
// K8: layernorm of sr (b,n,c) over c -> bf16 (b,n,c)
// ---------------------------------------------------------------------------
__global__ void ln_sr_kernel(const float* __restrict__ sr,
                             const float* __restrict__ g,
                             const float* __restrict__ bta,
                             __bf16* __restrict__ outB) {
  int row = blockIdx.x * 256 + threadIdx.x;
  size_t off = (size_t)row * 64;
  float sum = 0.f, sq = 0.f;
  for (int c = 0; c < 64; ++c) { float v = sr[off + c]; sum += v; sq += v * v; }
  float mean = sum * (1.0f / 64.0f);
  float var  = sq * (1.0f / 64.0f) - mean * mean;
  float inv  = rsqrtf(var + 1e-5f);
  for (int c = 0; c < 64; ++c)
    outB[off + c] = (__bf16)((sr[off + c] - mean) * inv * g[c] + bta[c]);
}

// ---------------------------------------------------------------------------
// K9: depthwise 3x3 + leaky(0.1), input f32 (b,n,c), output bf16 (b,n,c)
// ---------------------------------------------------------------------------
__global__ void dw_kernel(const float* __restrict__ t1,
                          const float* __restrict__ dwW,   // [64][9]
                          __bf16* __restrict__ y2) {
  int tid = blockIdx.x * 256 + threadIdx.x;        // over NTOT*64
  int c = tid & 63;
  int rowg = tid >> 6;
  int b = rowg >> 16, n = rowg & 65535;
  int iw = n & 255, ih = n >> 8;
  float s = 0.f;
#pragma unroll
  for (int a = 0; a < 3; ++a) {
#pragma unroll
    for (int q = 0; q < 3; ++q) {
      int iw2 = iw + a - 1, ih2 = ih + q - 1;
      if (iw2 >= 0 && iw2 < 256 && ih2 >= 0 && ih2 < 256) {
        int n2 = (ih2 << 8) | iw2;
        s += dwW[c * 9 + a * 3 + q] * t1[(size_t)((b << 16) | n2) * 64 + c];
      }
    }
  }
  s = (s > 0.f) ? s : 0.1f * s;
  y2[tid] = (__bf16)s;
}

// ---------------------------------------------------------------------------
// Host launcher
// ---------------------------------------------------------------------------
extern "C" void kernel_launch(void* const* d_in, const int* in_sizes, int n_in,
                              void* d_out, int out_size, void* d_ws, size_t ws_size,
                              hipStream_t stream) {
  const float* x_msi  = (const float*)d_in[0];
  const float* x_hsi  = (const float*)d_in[1];
  const float* lnm_g  = (const float*)d_in[2];
  const float* lnm_b  = (const float*)d_in[3];
  const float* lnh_g  = (const float*)d_in[4];
  const float* lnh_b  = (const float*)d_in[5];
  const float* lno_g  = (const float*)d_in[6];
  const float* lno_b  = (const float*)d_in[7];
  const float* wq     = (const float*)d_in[8];
  const float* wk     = (const float*)d_in[9];
  const float* wv     = (const float*)d_in[10];
  const float* w_proj = (const float*)d_in[11];
  const float* b_proj = (const float*)d_in[12];
  const float* pos_w1 = (const float*)d_in[13];
  const float* pos_b1 = (const float*)d_in[14];
  const float* pos_w2 = (const float*)d_in[15];
  const float* pos_b2 = (const float*)d_in[16];
  const float* ffn_w1 = (const float*)d_in[17];
  const float* ffn_dw = (const float*)d_in[18];
  const float* ffn_w2 = (const float*)d_in[19];
  float* out = (float*)d_out;

  // --- workspace layout (bytes), with deliberate buffer reuse -------------
  const size_t F32SZ = (size_t)NTOT * 64 * 4;   // 67,108,864
  const size_t BFSZ  = (size_t)NTOT * 64 * 2;   // 33,554,432
  const size_t OFF_XMNF  = 0;                   // f32 LN(msi) NCHW; later t1 f32
  const size_t OFF_XMNBF = OFF_XMNF  + F32SZ;   // bf16 LN(msi) (b,n,c); later srn
  const size_t OFF_XHNBF = OFF_XMNBF + BFSZ;    // bf16 LN(hsi); later y2
  const size_t OFF_QBF   = OFF_XHNBF + BFSZ;    // bf16 q; later pbuf (spans q+k)
  const size_t OFF_KBF   = OFF_QBF   + BFSZ;    // bf16 k
  const size_t OFF_VBF   = OFF_KBF   + BFSZ;    // bf16 v
  const size_t OFF_STATS = OFF_VBF   + BFSZ;    // S(4096f) nk2(256f) nq2(256f) attn(4096f)
  const size_t OFF_WBF   = OFF_STATS + 65536;   // bf16 weight buffers (~197 KB)
  const size_t OFF_SR    = OFF_WBF   + 262144;  // f32 sr (b,n,c)
  const size_t WS_NEED   = OFF_SR    + F32SZ;   // ~302.4 MB
  if (ws_size < WS_NEED) return;

  char* ws = (char*)d_ws;
  float*  xmnf32 = (float*)(ws + OFF_XMNF);
  __bf16* xmnbf  = (__bf16*)(ws + OFF_XMNBF);
  __bf16* xhnbf  = (__bf16*)(ws + OFF_XHNBF);
  __bf16* qbf    = (__bf16*)(ws + OFF_QBF);
  __bf16* kbf    = (__bf16*)(ws + OFF_KBF);
  __bf16* vbf    = (__bf16*)(ws + OFF_VBF);
  float*  S      = (float*)(ws + OFF_STATS);
  float*  nk2    = S + 4096;
  float*  nq2    = S + 4352;
  float*  attn   = S + 4608;
  float*  srbuf  = (float*)(ws + OFF_SR);
  float*  pbuf   = (float*)(ws + OFF_QBF);      // reuses q+k after gram
  float*  t1     = (float*)(ws + OFF_XMNF);     // reuses xmnf32 after conv1
  __bf16* srnbf  = (__bf16*)(ws + OFF_XMNBF);   // reuses xmnbf after qkv
  __bf16* y2bf   = (__bf16*)(ws + OFF_XHNBF);   // reuses xhnbf after qkv

  // bf16 weight buffers (fragment-friendly layouts)
  __bf16* wb     = (__bf16*)(ws + OFF_WBF);
  __bf16* wqT    = wb;            // [n][k]
  __bf16* wkT    = wb + 4096;
  __bf16* wvT    = wb + 8192;
  __bf16* wprojT = wb + 12288;
  __bf16* ffn1B  = wb + 16384;    // already [n=co][k=ci]
  __bf16* ffn2B  = wb + 20480;
  __bf16* posw1B = wb + 24576;    // [co][576]
  __bf16* posw2B = wb + 61440;

  hipMemsetAsync(S, 0, 4608 * sizeof(float), stream);   // zero S, nk2, nq2

  // 0) weight prep (tiny)
  cvt_t64_kernel<<<16, 256, 0, stream>>>(wq, wqT);
  cvt_t64_kernel<<<16, 256, 0, stream>>>(wk, wkT);
  cvt_t64_kernel<<<16, 256, 0, stream>>>(wv, wvT);
  cvt_t64_kernel<<<16, 256, 0, stream>>>(w_proj, wprojT);
  cvt_copy_kernel<<<16, 256, 0, stream>>>(ffn_w1, ffn1B, 4096);
  cvt_copy_kernel<<<16, 256, 0, stream>>>(ffn_w2, ffn2B, 4096);
  cvt_copy_kernel<<<144, 256, 0, stream>>>(pos_w1, posw1B, 36864);
  cvt_copy_kernel<<<144, 256, 0, stream>>>(pos_w2, posw2B, 36864);

  // 1) LN + transpose
  ln_tr_kernel<<<NTOT / 256, 256, 0, stream>>>(x_msi, lnm_g, lnm_b, xmnf32, xmnbf);
  ln_tr_kernel<<<NTOT / 256, 256, 0, stream>>>(x_hsi, lnh_g, lnh_b, nullptr, xhnbf);

  // 2) QKV projections (WMMA)
  gemm_n64_kernel<<<NTOT / 64, 128, 0, stream>>>(xhnbf, wqT, 0, 0, nullptr, qbf, nullptr, nullptr);
  gemm_n64_kernel<<<NTOT / 64, 128, 0, stream>>>(xmnbf, wkT, 0, 0, nullptr, kbf, nullptr, nullptr);
  gemm_n64_kernel<<<NTOT / 64, 128, 0, stream>>>(xmnbf, wvT, 0, 0, nullptr, vbf, nullptr, nullptr);

  // 3) norms + Gram + softmax (folded L2-normalization)
  norms_kernel<<<512, 256, 0, stream>>>(kbf, qbf, nk2, nq2);
  gram_kernel<<<dim3(64, 16), 256, 0, stream>>>(kbf, qbf, S);
  softmax_kernel<<<1, 256, 0, stream>>>(S, nk2, nq2, attn);

  // 4) xo = attn@v, out_c = gelu(xo@w_proj + b_proj) -> sr buffer (WMMA)
  av_proj_kernel<<<NTOT / 64, 128, 0, stream>>>(vbf, attn, wprojT, b_proj, srbuf);

  // 5) positional conv path (implicit-GEMM WMMA, LDS-staged input)
  conv3x3_kernel<<<NTOT / 64, 128, 0, stream>>>(xmnf32, posw1B, pos_b1, 1, 0,
                                                pbuf, nullptr, nullptr);
  conv3x3_kernel<<<NTOT / 64, 128, 0, stream>>>(pbuf, posw2B, pos_b2, 0, 1,
                                                nullptr, srbuf, x_msi);

  // 6) LN(sr) -> bf16, FFN: 1x1 -> leaky -> dw3x3 -> leaky -> 1x1 + residual
  ln_sr_kernel<<<NTOT / 256, 256, 0, stream>>>(srbuf, lno_g, lno_b, srnbf);
  gemm_n64_kernel<<<NTOT / 64, 128, 0, stream>>>(srnbf, ffn1B, 2, 1, t1, nullptr, nullptr, nullptr);
  dw_kernel<<<(NTOT * 64) / 256, 256, 0, stream>>>(t1, ffn_dw, y2bf);
  gemm_n64_kernel<<<NTOT / 64, 128, 0, stream>>>(y2bf, ffn2B, 0, 2, nullptr, nullptr, srbuf, out);

  (void)in_sizes; (void)n_in; (void)out_size;
}
